// AxialAttentionBlock_27281632264735
// MI455X (gfx1250) — compile-verified, matrix operand order checked by reading, not run on hardware
//
#include <hip/hip_runtime.h>
#include <math.h>

typedef __attribute__((ext_vector_type(16))) __bf16 bf16x16;
typedef __attribute__((ext_vector_type(8)))  __bf16 bf16x8;
typedef __attribute__((ext_vector_type(8)))  float  f32x8;

#define HEADS 12
#define C_    768
#define HD    64
#define HW_   4096
#define B_    8
#define M_    32768   /* B_*HW_ */

// ---------------------------------------------------------------------------
// WMMA helpers
// ---------------------------------------------------------------------------
__device__ __forceinline__ f32x8 wmma_bf16(bf16x16 a, bf16x16 b, f32x8 c) {
  return __builtin_amdgcn_wmma_f32_16x16x32_bf16(false, a, false, b, (short)0, c,
                                                 false, false);
}

// Load a 16x32 bf16 A-fragment (or B^T fragment) where the K dimension is
// contiguous in memory. Matches the CDNA5 16-bit A layout:
//   lanes 0-15 : row = lane,    K = kk+[0..7],  kk+[16..23]
//   lanes 16-31: row = lane-16, K = kk+[8..15], kk+[24..31]
__device__ __forceinline__ bf16x16 load_frag(const __bf16* base, int ld,
                                             int row0, int kk, int lane) {
  int half = (lane >> 4) & 1;
  int r    = lane & 15;
  const __bf16* p = base + (size_t)(row0 + r) * (size_t)ld + kk + half * 8;
  bf16x8 lo = *(const bf16x8*)(p);
  bf16x8 hi = *(const bf16x8*)(p + 16);
  bf16x16 f;
#pragma unroll
  for (int i = 0; i < 8; ++i) { f[i] = lo[i]; f[i + 8] = hi[i]; }
  return f;
}

// 32(M) x 64(N) per-wave GEMM mainloop, f32 accumulate.
// A: [M,K] row-major bf16,  Bt: [N,K] row-major bf16 (i.e. B^T, K contiguous)
__device__ __forceinline__ void gemm_core(const __bf16* __restrict__ A,
                                          const __bf16* __restrict__ Bt, int K,
                                          int row0, int col0, int lane,
                                          f32x8 acc[8]) {
  for (int kk = 0; kk < K; kk += 32) {
    __builtin_prefetch(A + (size_t)row0 * K + kk + 128, 0, 1);
    bf16x16 a0 = load_frag(A, K, row0,      kk, lane);
    bf16x16 a1 = load_frag(A, K, row0 + 16, kk, lane);
#pragma unroll
    for (int ni = 0; ni < 4; ++ni) {
      bf16x16 b = load_frag(Bt, K, col0 + ni * 16, kk, lane);
      acc[ni]     = wmma_bf16(a0, b, acc[ni]);
      acc[4 + ni] = wmma_bf16(a1, b, acc[4 + ni]);
    }
  }
}

// ---------------------------------------------------------------------------
// Weight preparation
// ---------------------------------------------------------------------------
__global__ void f32_to_bf16_kernel(const float* __restrict__ s,
                                   __bf16* __restrict__ d, int n) {
  int i = blockIdx.x * 256 + threadIdx.x;
  if (i < n) d[i] = (__bf16)s[i];
}

// d[n*K + k] = s[k*N + n]   (make K contiguous for B^T fragments)
__global__ void transpose_to_bf16_kernel(const float* __restrict__ s,
                                         __bf16* __restrict__ d, int K, int N) {
  int i = blockIdx.x * 256 + threadIdx.x;
  if (i < K * N) {
    int n = i / K, k = i - n * K;
    d[i] = (__bf16)s[(size_t)k * N + n];
  }
}

// ---------------------------------------------------------------------------
// RMS instance norm (std over H*W, unbiased), x layout [B,C,HW] contiguous
// ---------------------------------------------------------------------------
__global__ __launch_bounds__(256) void inorm_chw_kernel(
    const float* __restrict__ x, const float* __restrict__ w,
    float* __restrict__ s) {
  int bc = blockIdx.x, tid = threadIdx.x;
  const float* p = x + (size_t)bc * HW_;
  float sum = 0.f, sq = 0.f;
  for (int i = tid; i < HW_; i += 256) { float v = p[i]; sum += v; sq += v * v; }
  __shared__ float rs[256], rq[256];
  rs[tid] = sum; rq[tid] = sq; __syncthreads();
  for (int st = 128; st > 0; st >>= 1) {
    if (tid < st) { rs[tid] += rs[tid + st]; rq[tid] += rq[tid + st]; }
    __syncthreads();
  }
  if (tid == 0) {
    float mean = rs[0] * (1.f / HW_);
    float var  = (rq[0] - (float)HW_ * mean * mean) * (1.f / (HW_ - 1));
    s[bc] = w[bc % C_] / (sqrtf(var) + 1e-6f);
  }
}

// same, but for token-major tensors a[m, c]; s[b*C+c] = w[c]/(std+eps)
__global__ __launch_bounds__(256) void inorm_tok_kernel(
    const float* __restrict__ a, const float* __restrict__ w,
    float* __restrict__ s) {
  int cb = blockIdx.x * 32, b = blockIdx.y;
  int ci = threadIdx.x & 31, r = threadIdx.x >> 5;
  float sum = 0.f, sq = 0.f;
  for (int hw = r; hw < HW_; hw += 8) {
    float v = a[((size_t)b * HW_ + hw) * C_ + cb + ci];
    sum += v; sq += v * v;
  }
  __shared__ float rs[256], rq[256];
  rs[threadIdx.x] = sum; rq[threadIdx.x] = sq; __syncthreads();
  if (r == 0) {
#pragma unroll
    for (int g = 1; g < 8; ++g) { sum += rs[g * 32 + ci]; sq += rq[g * 32 + ci]; }
    float mean = sum * (1.f / HW_);
    float var  = (sq - (float)HW_ * mean * mean) * (1.f / (HW_ - 1));
    s[b * C_ + cb + ci] = w[cb + ci] / (sqrtf(var) + 1e-6f);
  }
}

// h[b, hw, c] = bf16(x[b, c, hw] * s[b, c])   (LDS-tiled 32x32 transpose)
__global__ __launch_bounds__(256) void transpose_scale_kernel(
    const float* __restrict__ x, const float* __restrict__ s,
    __bf16* __restrict__ h) {
  __shared__ float t[32][33];
  int hw0 = blockIdx.x * 32, c0 = blockIdx.y * 32, b = blockIdx.z;
  int li = threadIdx.x & 31, gr = threadIdx.x >> 5;
#pragma unroll
  for (int rr = 0; rr < 4; ++rr) {
    int c = c0 + gr + rr * 8;
    t[gr + rr * 8][li] = x[((size_t)b * C_ + c) * HW_ + hw0 + li] * s[b * C_ + c];
  }
  __syncthreads();
#pragma unroll
  for (int rr = 0; rr < 4; ++rr) {
    int hw = hw0 + gr + rr * 8;
    h[((size_t)b * HW_ + hw) * C_ + c0 + li] = (__bf16)t[li][gr + rr * 8];
  }
}

// d[i] = bf16(a[i] * s[b, c])   token-major elementwise
__global__ void scale_convert_kernel(const float* __restrict__ a,
                                     const float* __restrict__ s,
                                     __bf16* __restrict__ d) {
  int i = blockIdx.x * 256 + threadIdx.x;
  int c = i % C_;
  int m = i / C_;
  int b = m >> 12;
  d[i] = (__bf16)(a[i] * s[b * C_ + c]);
}

// ---------------------------------------------------------------------------
// QKV GEMM: qkv[m, o] = h[m,:] . Win[o,:] + bin[o]; scatter to q/k/v bf16
// ---------------------------------------------------------------------------
__global__ __launch_bounds__(256) void gemm_qkv_kernel(
    const __bf16* __restrict__ A, const __bf16* __restrict__ Bt,
    const float* __restrict__ bias, __bf16* __restrict__ qd,
    __bf16* __restrict__ kd, __bf16* __restrict__ vd) {
  int tid = threadIdx.x, lane = tid & 31, wv = tid >> 5;
  int row0 = blockIdx.x * 64 + (wv & 1) * 32;
  int col0 = blockIdx.y * 256 + (wv >> 1) * 64;
  const f32x8 zero = {0.f, 0.f, 0.f, 0.f, 0.f, 0.f, 0.f, 0.f};
  f32x8 acc[8];
#pragma unroll
  for (int i = 0; i < 8; ++i) acc[i] = zero;
  gemm_core(A, Bt, C_, row0, col0, lane, acc);
  int half = lane >> 4, r = lane & 15;
#pragma unroll
  for (int mi = 0; mi < 2; ++mi)
#pragma unroll
    for (int ni = 0; ni < 4; ++ni)
#pragma unroll
      for (int j = 0; j < 8; ++j) {
        int row = row0 + mi * 16 + j + 8 * half;
        int col = col0 + ni * 16 + r;
        float val = acc[mi * 4 + ni][j] + bias[col];
        int b = row >> 12, hw = row & 4095;
        int he = col / (3 * HD), rem = col - he * 3 * HD;
        int t = rem >> 6, d = rem & 63;
        __bf16* dst = (t == 0) ? qd : (t == 1) ? kd : vd;
        dst[(((size_t)(b * HEADS + he)) * HW_ + hw) * HD + d] = (__bf16)val;
      }
}

// Wout GEMM + layerscale + residual: x1 = (a.Wout^T + bout)*gamma + x
__global__ __launch_bounds__(256) void gemm_wout_kernel(
    const __bf16* __restrict__ A, const __bf16* __restrict__ Bt,
    const float* __restrict__ bias, const float* __restrict__ gatt,
    const float* __restrict__ x, float* __restrict__ x1f,
    __bf16* __restrict__ x1b) {
  int tid = threadIdx.x, lane = tid & 31, wv = tid >> 5;
  int row0 = blockIdx.x * 64 + (wv & 1) * 32;
  int col0 = blockIdx.y * 256 + (wv >> 1) * 64;
  const f32x8 zero = {0.f, 0.f, 0.f, 0.f, 0.f, 0.f, 0.f, 0.f};
  f32x8 acc[8];
#pragma unroll
  for (int i = 0; i < 8; ++i) acc[i] = zero;
  gemm_core(A, Bt, C_, row0, col0, lane, acc);
  int half = lane >> 4, r = lane & 15;
#pragma unroll
  for (int mi = 0; mi < 2; ++mi)
#pragma unroll
    for (int ni = 0; ni < 4; ++ni)
#pragma unroll
      for (int j = 0; j < 8; ++j) {
        int row = row0 + mi * 16 + j + 8 * half;
        int col = col0 + ni * 16 + r;
        int b = row >> 12, hw = row & 4095;
        float val = acc[mi * 4 + ni][j] + bias[col];
        float xv  = x[((size_t)b * C_ + col) * HW_ + hw];
        float x1  = fmaf(val, gatt[col], xv);
        size_t oi = (size_t)row * C_ + col;
        x1f[oi] = x1;
        x1b[oi] = (__bf16)x1;
      }
}

// MLP1: g = gelu(x1.W1 + b1)  (exact erf gelu), bf16 out [M, 4C]
__global__ __launch_bounds__(256) void gemm_mlp1_kernel(
    const __bf16* __restrict__ A, const __bf16* __restrict__ Bt,
    const float* __restrict__ bias, __bf16* __restrict__ gb) {
  int tid = threadIdx.x, lane = tid & 31, wv = tid >> 5;
  int row0 = blockIdx.x * 64 + (wv & 1) * 32;
  int col0 = blockIdx.y * 256 + (wv >> 1) * 64;
  const f32x8 zero = {0.f, 0.f, 0.f, 0.f, 0.f, 0.f, 0.f, 0.f};
  f32x8 acc[8];
#pragma unroll
  for (int i = 0; i < 8; ++i) acc[i] = zero;
  gemm_core(A, Bt, C_, row0, col0, lane, acc);
  int half = lane >> 4, r = lane & 15;
#pragma unroll
  for (int mi = 0; mi < 2; ++mi)
#pragma unroll
    for (int ni = 0; ni < 4; ++ni)
#pragma unroll
      for (int j = 0; j < 8; ++j) {
        int row = row0 + mi * 16 + j + 8 * half;
        int col = col0 + ni * 16 + r;
        float u = acc[mi * 4 + ni][j] + bias[col];
        float g = 0.5f * u * (1.f + erff(u * 0.70710678118654752f));
        gb[(size_t)row * (4 * C_) + col] = (__bf16)g;
      }
}

// MLP2: m = g.W2 + b2, f32 out [M, C]
__global__ __launch_bounds__(256) void gemm_mlp2_kernel(
    const __bf16* __restrict__ A, const __bf16* __restrict__ Bt,
    const float* __restrict__ bias, float* __restrict__ mo) {
  int tid = threadIdx.x, lane = tid & 31, wv = tid >> 5;
  int row0 = blockIdx.x * 64 + (wv & 1) * 32;
  int col0 = blockIdx.y * 256 + (wv >> 1) * 64;
  const f32x8 zero = {0.f, 0.f, 0.f, 0.f, 0.f, 0.f, 0.f, 0.f};
  f32x8 acc[8];
#pragma unroll
  for (int i = 0; i < 8; ++i) acc[i] = zero;
  gemm_core(A, Bt, 4 * C_, row0, col0, lane, acc);
  int half = lane >> 4, r = lane & 15;
#pragma unroll
  for (int mi = 0; mi < 2; ++mi)
#pragma unroll
    for (int ni = 0; ni < 4; ++ni)
#pragma unroll
      for (int j = 0; j < 8; ++j) {
        int row = row0 + mi * 16 + j + 8 * half;
        int col = col0 + ni * 16 + r;
        mo[(size_t)row * C_ + col] = acc[mi * 4 + ni][j] + bias[col];
      }
}

// ---------------------------------------------------------------------------
// Per-head LayerNorm over hd=64, in-place on bf16 q or k
// ---------------------------------------------------------------------------
__global__ __launch_bounds__(256) void ln_head_kernel(
    __bf16* __restrict__ t, const float* __restrict__ w,
    const float* __restrict__ bias, int ntok) {
  int lane = threadIdx.x & 31;
  int grp = blockIdx.x * 8 + (threadIdx.x >> 5);
  if (grp >= ntok) return;
  __bf16* p = t + (size_t)grp * HD;
  float v0 = (float)p[lane], v1 = (float)p[lane + 32];
  float sm = v0 + v1;
#pragma unroll
  for (int d = 1; d < 32; d <<= 1) sm += __shfl_xor(sm, d, 32);
  float mean = sm * (1.f / HD);
  float d0 = v0 - mean, d1 = v1 - mean;
  float q = d0 * d0 + d1 * d1;
#pragma unroll
  for (int d = 1; d < 32; d <<= 1) q += __shfl_xor(q, d, 32);
  float rstd = rsqrtf(q * (1.f / HD) + 1e-5f);
  p[lane]      = (__bf16)(d0 * rstd * w[lane] + bias[lane]);
  p[lane + 32] = (__bf16)(d1 * rstd * w[lane + 32] + bias[lane + 32]);
}

// ---------------------------------------------------------------------------
// Axial attention, seqlen=64, hd=64, one block per (b, head, row|col).
// colmode==0: attend along W (writes raw xx into a_tok)
// colmode==1: attend along H (reads xx, writes 0.5*(xx+xy))
// ---------------------------------------------------------------------------
__global__ __launch_bounds__(128) void attn_kernel(
    const __bf16* __restrict__ q, const __bf16* __restrict__ k,
    const __bf16* __restrict__ v, float* __restrict__ out, int colmode) {
  int bid = blockIdx.x;
  int s  = bid & 63;
  int he = (bid >> 6) % HEADS;
  int b  = bid / (HEADS * 64);
  size_t head = ((size_t)(b * HEADS + he)) * HW_ * HD;
  int ld = colmode ? (64 * HD) : HD;
  size_t base = head + (colmode ? (size_t)s * HD : (size_t)s * 64 * HD);
  const __bf16* qb = q + base;
  const __bf16* kb = k + base;
  const __bf16* vb = v + base;
  float* ob = out + ((size_t)b * HW_ + (colmode ? s : s * 64)) * C_ + he * HD;
  size_t ldo = colmode ? (size_t)64 * C_ : (size_t)C_;

  __shared__ __bf16 Vt[64 * 72];  // V transposed: [hd][key], padded rows
  __shared__ __bf16 P[64 * 72];   // softmax probs, padded rows

  int tid = threadIdx.x, lane = tid & 31, wv = tid >> 5;

  for (int idx = tid; idx < 64 * HD; idx += 128) {
    int key = idx >> 6, d = idx & 63;
    Vt[d * 72 + key] = vb[(size_t)key * ld + d];
  }
  __syncthreads();

  int m0 = wv * 16;  // 16 query rows per wave
  const f32x8 zero = {0.f, 0.f, 0.f, 0.f, 0.f, 0.f, 0.f, 0.f};
  f32x8 sacc[4] = {zero, zero, zero, zero};

  bf16x16 qf0 = load_frag(qb, ld, m0, 0, lane);
  bf16x16 qf1 = load_frag(qb, ld, m0, 32, lane);
#pragma unroll
  for (int nt = 0; nt < 4; ++nt) {
    bf16x16 k0 = load_frag(kb, ld, nt * 16, 0, lane);
    bf16x16 k1 = load_frag(kb, ld, nt * 16, 32, lane);
    sacc[nt] = wmma_bf16(qf0, k0, sacc[nt]);
    sacc[nt] = wmma_bf16(qf1, k1, sacc[nt]);
  }

  const float scale = 0.125f;  // 1/sqrt(64)
  int half = lane >> 4, r = lane & 15;
  float rinv[8];
#pragma unroll
  for (int j = 0; j < 8; ++j) {
    float mx = -3.0e38f;
#pragma unroll
    for (int nt = 0; nt < 4; ++nt) {
      sacc[nt][j] *= scale;
      mx = fmaxf(mx, sacc[nt][j]);
    }
#pragma unroll
    for (int d = 1; d < 16; d <<= 1) mx = fmaxf(mx, __shfl_xor(mx, d, 32));
    float sum = 0.f, pj[4];
#pragma unroll
    for (int nt = 0; nt < 4; ++nt) {
      pj[nt] = __expf(sacc[nt][j] - mx);
      sum += pj[nt];
    }
#pragma unroll
    for (int d = 1; d < 16; d <<= 1) sum += __shfl_xor(sum, d, 32);
    rinv[j] = 1.f / sum;
    int prow = m0 + j + 8 * half;
#pragma unroll
    for (int nt = 0; nt < 4; ++nt) P[prow * 72 + nt * 16 + r] = (__bf16)pj[nt];
  }
  __syncthreads();

  bf16x16 p0 = load_frag(P, 72, m0, 0, lane);
  bf16x16 p1 = load_frag(P, 72, m0, 32, lane);
#pragma unroll
  for (int nt2 = 0; nt2 < 4; ++nt2) {
    f32x8 o = zero;
    bf16x16 v0 = load_frag(Vt, 72, nt2 * 16, 0, lane);
    bf16x16 v1 = load_frag(Vt, 72, nt2 * 16, 32, lane);
    o = wmma_bf16(p0, v0, o);
    o = wmma_bf16(p1, v1, o);
#pragma unroll
    for (int j = 0; j < 8; ++j) {
      float val = o[j] * rinv[j];
      int row = m0 + j + 8 * half;
      size_t oi = (size_t)row * ldo + nt2 * 16 + r;
      if (colmode) ob[oi] = 0.5f * (ob[oi] + val);
      else         ob[oi] = val;
    }
  }
}

// final: out[b,c,hw] = x1[m,c] + gamma_mlp[c]*s3[b,c]*m_raw[m,c]
__global__ __launch_bounds__(256) void final_kernel(
    const float* __restrict__ x1, const float* __restrict__ m,
    const float* __restrict__ s3, const float* __restrict__ gm,
    float* __restrict__ out) {
  int bc = blockIdx.x;
  int b = bc / C_, c = bc - b * C_;
  float g = gm[c] * s3[bc];
  for (int hw = threadIdx.x; hw < HW_; hw += 256) {
    size_t ti = ((size_t)b * HW_ + hw) * C_ + c;
    out[(size_t)bc * HW_ + hw] = x1[ti] + g * m[ti];
  }
}

// ---------------------------------------------------------------------------
extern "C" void kernel_launch(void* const* d_in, const int* in_sizes, int n_in,
                              void* d_out, int out_size, void* d_ws,
                              size_t ws_size, hipStream_t stream) {
  const float* x       = (const float*)d_in[0];
  const float* norm1_w = (const float*)d_in[2];
  const float* norm2_w = (const float*)d_in[3];
  const float* mlpn_w  = (const float*)d_in[4];
  const float* Win     = (const float*)d_in[5];
  const float* bin_    = (const float*)d_in[6];
  const float* Wout    = (const float*)d_in[7];
  const float* bout    = (const float*)d_in[8];
  const float* qn_w    = (const float*)d_in[9];
  const float* qn_b    = (const float*)d_in[10];
  const float* kn_w    = (const float*)d_in[11];
  const float* kn_b    = (const float*)d_in[12];
  const float* gatt    = (const float*)d_in[13];
  const float* gmlp    = (const float*)d_in[14];
  const float* W1      = (const float*)d_in[15];
  const float* b1      = (const float*)d_in[16];
  const float* W2      = (const float*)d_in[17];
  const float* b2      = (const float*)d_in[18];
  float* out = (float*)d_out;

  char* wsb = (char*)d_ws;
  size_t off = 0;
  auto alloc = [&](size_t bytes) -> void* {
    void* p = wsb + off;
    off = (off + bytes + 255) & ~(size_t)255;
    return p;
  };
  float* s1 = (float*)alloc((size_t)B_ * C_ * 4);
  float* s2 = (float*)alloc((size_t)B_ * C_ * 4);
  float* s3 = (float*)alloc((size_t)B_ * C_ * 4);
  __bf16* Win_bf  = (__bf16*)alloc((size_t)3 * C_ * C_ * 2);
  __bf16* Wout_bf = (__bf16*)alloc((size_t)C_ * C_ * 2);
  __bf16* W1t     = (__bf16*)alloc((size_t)4 * C_ * C_ * 2);
  __bf16* W2t     = (__bf16*)alloc((size_t)4 * C_ * C_ * 2);
  __bf16* h_bf    = (__bf16*)alloc((size_t)M_ * C_ * 2);      // reused as abf
  __bf16* qd      = (__bf16*)alloc((size_t)M_ * C_ * 2);
  __bf16* kd      = (__bf16*)alloc((size_t)M_ * C_ * 2);
  __bf16* vd      = (__bf16*)alloc((size_t)M_ * C_ * 2);      // reused as x1_bf
  float*  a_tok   = (float*)alloc((size_t)M_ * C_ * 4);       // reused as m_raw
  float*  x1_tok  = (float*)alloc((size_t)M_ * C_ * 4);
  __bf16* g_bf    = (__bf16*)alloc((size_t)M_ * 4 * C_ * 2);
  __bf16* abf   = h_bf;   // h dead after QKV gemm
  __bf16* x1_bf = vd;     // v dead after attention
  float*  m_raw = a_tok;  // a dead after Wout gemm

  // weight prep (bf16, K-contiguous B^T)
  f32_to_bf16_kernel<<<(3 * C_ * C_ + 255) / 256, 256, 0, stream>>>(
      Win, Win_bf, 3 * C_ * C_);
  f32_to_bf16_kernel<<<(C_ * C_ + 255) / 256, 256, 0, stream>>>(
      Wout, Wout_bf, C_ * C_);
  transpose_to_bf16_kernel<<<(4 * C_ * C_ + 255) / 256, 256, 0, stream>>>(
      W1, W1t, C_, 4 * C_);
  transpose_to_bf16_kernel<<<(4 * C_ * C_ + 255) / 256, 256, 0, stream>>>(
      W2, W2t, 4 * C_, C_);

  // norm1 + tokenize
  inorm_chw_kernel<<<B_ * C_, 256, 0, stream>>>(x, norm1_w, s1);
  transpose_scale_kernel<<<dim3(HW_ / 32, C_ / 32, B_), 256, 0, stream>>>(
      x, s1, h_bf);

  // QKV
  gemm_qkv_kernel<<<dim3(M_ / 64, (3 * C_) / 256), 256, 0, stream>>>(
      h_bf, Win_bf, bin_, qd, kd, vd);
  int ntok = B_ * HEADS * HW_;
  ln_head_kernel<<<ntok / 8, 256, 0, stream>>>(qd, qn_w, qn_b, ntok);
  ln_head_kernel<<<ntok / 8, 256, 0, stream>>>(kd, kn_w, kn_b, ntok);

  // axial attention (row then column, column merges with 0.5*(xx+xy))
  attn_kernel<<<B_ * HEADS * 64, 128, 0, stream>>>(qd, kd, vd, a_tok, 0);
  attn_kernel<<<B_ * HEADS * 64, 128, 0, stream>>>(qd, kd, vd, a_tok, 1);

  // norm2 + proj + layerscale + residual
  inorm_tok_kernel<<<dim3(C_ / 32, B_), 256, 0, stream>>>(a_tok, norm2_w, s2);
  scale_convert_kernel<<<(M_ * C_) / 256, 256, 0, stream>>>(a_tok, s2, abf);
  gemm_wout_kernel<<<dim3(M_ / 64, C_ / 256), 256, 0, stream>>>(
      abf, Wout_bf, bout, gatt, x, x1_tok, x1_bf);

  // MLP
  gemm_mlp1_kernel<<<dim3(M_ / 64, (4 * C_) / 256), 256, 0, stream>>>(
      x1_bf, W1t, b1, g_bf);
  gemm_mlp2_kernel<<<dim3(M_ / 64, C_ / 256), 256, 0, stream>>>(
      g_bf, W2t, b2, m_raw);

  // mlp norm + final residual
  inorm_tok_kernel<<<dim3(C_ / 32, B_), 256, 0, stream>>>(m_raw, mlpn_w, s3);
  final_kernel<<<B_ * C_, 256, 0, stream>>>(x1_tok, m_raw, s3, gmlp, out);

  (void)in_sizes; (void)n_in; (void)out_size; (void)ws_size;
}
